// MoMAttentionCross_78391743086628
// MI455X (gfx1250) — compile-verified, matrix operand order checked by reading, not run on hardware
//
#include <hip/hip_runtime.h>
#include <hip/hip_bf16.h>

// ---------------------------------------------------------------------------
// MoM cross-attention, MI455X (gfx1250, wave32, WMMA bf16 16x16x32)
//
// Sizes (from reference): B=8, Q=64, K=256, D=256, H=4, DK=DV=64, M=4, TOPK=2
// N = B*Q = 512 independent rows; one 256-thread workgroup (8 wave32) per row.
// ---------------------------------------------------------------------------

#define BB   8
#define QQ   64
#define KK   256
#define DD   256
#define HH   4
#define DKK  64
#define DVV  64
#define MMX  4
#define NN   (BB * QQ)      // 512
#define TT   16             // t-tile rows
#define NTILE (KK / TT)     // 16
#define NMEM 5              // base memory + 4 experts
#define GN_F 16.0f
#define EPS_F 1e-5f
#define COND_SCALE_F 1.0f

typedef __attribute__((ext_vector_type(16))) __bf16 v16bf;
typedef __attribute__((ext_vector_type(8)))  __bf16 v8bf;
typedef __attribute__((ext_vector_type(8)))  float  v8f;

// ---- workspace layout (bf16 elements) -------------------------------------
// W_k fragments : [0       , 65536)   8 kc * 16 j * 32 lane * 16 elem
// W_v fragments : [65536   , 131072)
// W_gk1 frags   : [131072  , 135168)  8 kc * 1 j * 512
// W_gk2 frags   : [135168  , 143360)  1 kc * 16 j * 512 (K padded 16->32 w/ 0)
#define WS_WK  0
#define WS_WV  65536
#define WS_G1  131072
#define WS_G2  135168
#define WS_TOT 143360

// ---- LDS layout (bytes) ----------------------------------------------------
#define L_XF    0        // float  [16][256]
#define L_XBF   16384    // __bf16 [16][256]
#define L_K     24576    // float  [16][256]
#define L_V     40960    // float  [16][256]
#define L_GK    57344    // float  [16][256]
#define L_WMEM  73728    // float  [256][4]   (full K, persists pass1->pass2)
#define L_COND  77824    // float  [256]
#define L_QL    78848    // float  [256]
#define L_G1BF  79872    // __bf16 [16][16]
#define L_RLOG  80384    // float  [16][4]
#define L_MT    80640    // float  [16]
#define L_SPART 80704    // float  [16][5][8]
#define L_RED   83264    // float  [8]
#define L_OFLAT 83296    // float  [256]
#define MAIN_SMEM 84480

static __device__ __forceinline__ __bf16 f2bf(float f) {
  union { float f; unsigned u; } v; v.f = f;
  unsigned r = v.u + 0x7FFFu + ((v.u >> 16) & 1u);   // round-to-nearest-even
  unsigned short s = (unsigned short)(r >> 16);
  __bf16 b; __builtin_memcpy(&b, &s, 2);
  return b;
}

static __device__ __forceinline__ v8f zero8() {
  v8f z;
#pragma unroll
  for (int i = 0; i < 8; ++i) z[i] = 0.0f;
  return z;
}

static __device__ __forceinline__ v8f wmma_bf16(v16bf a, v16bf b, v8f c) {
  // D(16x16,f32) = A(16x32,bf16) * B(32x16,bf16) + C
  return __builtin_amdgcn_wmma_f32_16x16x32_bf16(
      /*neg_a=*/false, a, /*neg_b=*/false, b,
      /*c_mod=*/(short)0, c, /*reuse_a=*/false, /*reuse_b=*/false);
}

// A fragment from a row-major LDS tile xbf[16][256], K-chunk kc (32 wide).
// ISA 16-bit A 16x32 layout: lane<16 holds K {0..7,16..23}, lane>=16 holds
// K {8..15,24..31}; row M = lane&15.  -> two 16B contiguous LDS reads.
static __device__ __forceinline__ v16bf lds_a_frag(const __bf16* xbf,
                                                   int lane, int kc) {
  const int t  = lane & 15;
  const int d0 = kc * 32 + ((lane >> 4) & 1) * 8;
  const v8bf* p0 = (const v8bf*)(xbf + t * DD + d0);
  const v8bf* p1 = (const v8bf*)(xbf + t * DD + d0 + 16);
  v16bf a;
#pragma unroll
  for (int i = 0; i < 8; ++i) { a[i] = (*p0)[i]; a[i + 8] = (*p1)[i]; }
  return a;
}

// A fragment for the 16x16 g1 matrix (effective K=16, upper half zero).
static __device__ __forceinline__ v16bf lds_a_frag_g1(const __bf16* g1bf,
                                                      int lane) {
  const int t    = lane & 15;
  const int base = ((lane >> 4) & 1) * 8;
  const __bf16 z = f2bf(0.0f);
  v16bf a;
#pragma unroll
  for (int i = 0; i < 8; ++i) { a[i] = g1bf[t * 16 + base + i]; a[i + 8] = z; }
  return a;
}

static __device__ __forceinline__ float wave_reduce(float v) {
#pragma unroll
  for (int m = 16; m >= 1; m >>= 1) v += __shfl_xor(v, m, 32);
  return v;
}

static __device__ __forceinline__ float log_sigmoid(float z) {
  return (z >= 0.0f) ? -log1pf(__expf(-z)) : (z - log1pf(__expf(z)));
}

// ---------------------------------------------------------------------------
// Kernel 1: pre-swizzle weights into CDNA5 bf16 B-fragment layout.
// Fragment element (lane,i): K_local = (i<8 ? kh*8+i : 16+kh*8+i-8),
// kh = lane>=16, N_local = lane&15.  Storage: ((kc*NJ+j)*32+lane)*16+i.
// ---------------------------------------------------------------------------
__global__ __launch_bounds__(256) void swizzle_weights(
    const float* __restrict__ Wk, const float* __restrict__ Wv,
    const float* __restrict__ Wgk1, const float* __restrict__ Wgk2,
    __bf16* __restrict__ ws) {
  const int g = blockIdx.x * 256 + threadIdx.x;
  if (g >= WS_TOT) return;

  int idx, region_base;
  const float* W;
  int ncols, k_max;
  if (g < WS_WV) {           // W_k
    idx = g;           region_base = WS_WK; W = Wk;   ncols = 256; k_max = 256;
  } else if (g < WS_G1) {    // W_v
    idx = g - WS_WV;   region_base = WS_WV; W = Wv;   ncols = 256; k_max = 256;
  } else if (g < WS_G2) {    // W_gk1 (256x16)
    idx = g - WS_G1;   region_base = WS_G1; W = Wgk1; ncols = 16;  k_max = 256;
  } else {                   // W_gk2 (16x256, K zero-padded to 32)
    idx = g - WS_G2;   region_base = WS_G2; W = Wgk2; ncols = 256; k_max = 16;
  }

  const int i    = idx & 15;
  const int lane = (idx >> 4) & 31;
  const int blk  = idx >> 9;                 // kc*NJ + j
  const int kh   = (lane >> 4) & 1;
  const int kl   = (i < 8) ? (kh * 8 + i) : (16 + kh * 8 + (i - 8));

  int kc, j;
  if (region_base == WS_G1)      { kc = blk; j = 0;   }
  else if (region_base == WS_G2) { kc = 0;   j = blk; }
  else                           { kc = blk >> 4; j = blk & 15; }

  const int k = kc * 32 + kl;
  const int n = j * 16 + (lane & 15);
  const float val = (k < k_max) ? W[k * ncols + n] : 0.0f;
  ws[region_base + idx] = f2bf(val);
}

// ---------------------------------------------------------------------------
// Kernel 2: full fused MoM attention for one row n = b*Q + q per workgroup.
// ---------------------------------------------------------------------------
__global__ __launch_bounds__(256) void mom_attn_kernel(
    const float* __restrict__ query, const float* __restrict__ keyval,
    const int* __restrict__ amask, const float* __restrict__ W_cond,
    const float* __restrict__ W_q, const float* __restrict__ b_gk2,
    const float* __restrict__ W_router, const float* __restrict__ norm_w,
    const float* __restrict__ W_o, const __bf16* __restrict__ ws,
    float* __restrict__ out, float* __restrict__ rlogits) {
  extern __shared__ char smem[];
  float*  xf    = (float*)(smem + L_XF);
  __bf16* xbf   = (__bf16*)(smem + L_XBF);
  float*  kls   = (float*)(smem + L_K);
  float*  vls   = (float*)(smem + L_V);
  float*  gkls  = (float*)(smem + L_GK);
  float*  wmem  = (float*)(smem + L_WMEM);
  float*  cond  = (float*)(smem + L_COND);
  float*  qlast = (float*)(smem + L_QL);
  __bf16* g1bf  = (__bf16*)(smem + L_G1BF);
  float*  rlog  = (float*)(smem + L_RLOG);
  float*  mt    = (float*)(smem + L_MT);
  float*  spart = (float*)(smem + L_SPART);
  float*  red   = (float*)(smem + L_RED);
  float*  oflat = (float*)(smem + L_OFLAT);

  const int tid  = threadIdx.x;
  const int lane = tid & 31;
  const int wave = tid >> 5;
  const int n    = blockIdx.x;
  const int b    = n >> 6;   // Q = 64

  const __bf16* wsWk = ws + WS_WK;
  const __bf16* wsWv = ws + WS_WV;
  const __bf16* wsG1 = ws + WS_G1;
  const __bf16* wsG2 = ws + WS_G2;

  // ---- cond row: query[n,:] @ W_cond -------------------------------------
  {
    const float* qr = query + (size_t)n * DD;
    float s = 0.0f;
#pragma unroll 8
    for (int d = 0; d < DD; ++d) s += qr[d] * W_cond[d * DD + tid];
    cond[tid] = s * COND_SCALE_F;
  }
  __syncthreads();

  float Gtot[NMEM];
#pragma unroll
  for (int m = 0; m < NMEM; ++m) Gtot[m] = 0.0f;

  // =========================================================================
  // PASS 1: gk + router/top-2 weights; accumulate per-memory total decay.
  // =========================================================================
  for (int tile = 0; tile < NTILE; ++tile) {
    const int bt = tile * TT;
    {
      const float* kv = keyval + ((size_t)(b * KK + bt)) * DD;
      const float cv = cond[tid];
#pragma unroll
      for (int i = 0; i < TT; ++i) {
        const float xv = kv[i * DD + tid] + cv;
        xf[i * DD + tid]  = xv;
        xbf[i * DD + tid] = f2bf(xv);
      }
      if (tid < TT) mt[tid] = (float)amask[b * KK + bt + tid];
    }
    __syncthreads();

    if (wave == 0) {  // g1 = x @ W_gk1 (16x16), fp32 acc
      v8f acc = zero8();
#pragma unroll
      for (int kc = 0; kc < 8; ++kc) {
        v16bf a = lds_a_frag(xbf, lane, kc);
        v16bf bb = *(const v16bf*)(wsG1 + ((size_t)(kc * 32 + lane)) * 16);
        acc = wmma_bf16(a, bb, acc);
      }
      const int rh = ((lane >> 4) & 1) * 8;
#pragma unroll
      for (int r = 0; r < 8; ++r)
        g1bf[(r + rh) * 16 + (lane & 15)] = f2bf(acc[r]);
    }
    if (tid >= 128 && tid < 192) {  // router logits in fp32 (exact path)
      const int t = (tid - 128) >> 2, m = (tid - 128) & 3;
      const int rot = (tid & 63) << 2;
      float s = 0.0f;
#pragma unroll 8
      for (int dd = 0; dd < DD; ++dd) {
        const int d = (dd + rot) & (DD - 1);
        s += xf[t * DD + d] * W_router[d * MMX + m];
      }
      rlog[t * MMX + m] = s;
      rlogits[((size_t)n * KK + bt + t) * MMX + m] = s;
    }
    __syncthreads();

    {  // gk = log_sigmoid(g1 @ W_gk2 + b) / GN, masked
      v16bf ag = lds_a_frag_g1(g1bf, lane);
#pragma unroll
      for (int jj = 0; jj < 2; ++jj) {
        const int j = wave * 2 + jj;
        v16bf bb = *(const v16bf*)(wsG2 + ((size_t)(j * 32 + lane)) * 16);
        v8f acc = wmma_bf16(ag, bb, zero8());
        const int nc = j * 16 + (lane & 15);
        const int rh = ((lane >> 4) & 1) * 8;
#pragma unroll
        for (int r = 0; r < 8; ++r) {
          const int tr = r + rh;
          const float z = acc[r] + b_gk2[nc];
          gkls[tr * DD + nc] = log_sigmoid(z) * (1.0f / GN_F) * mt[tr];
        }
      }
    }
    if (tid < TT) {  // softmax + top-2 + renormalize
      float l[MMX];
#pragma unroll
      for (int m = 0; m < MMX; ++m) l[m] = rlog[tid * MMX + m];
      float mx = l[0];
#pragma unroll
      for (int m = 1; m < MMX; ++m) mx = fmaxf(mx, l[m]);
      float p[MMX], ps = 0.0f;
#pragma unroll
      for (int m = 0; m < MMX; ++m) { p[m] = __expf(l[m] - mx); ps += p[m]; }
#pragma unroll
      for (int m = 0; m < MMX; ++m) p[m] /= ps;
      int i1 = 0;
#pragma unroll
      for (int m = 1; m < MMX; ++m) if (p[m] > p[i1]) i1 = m;
      int i2 = (i1 == 0) ? 1 : 0;
#pragma unroll
      for (int m = 0; m < MMX; ++m)
        if (m != i1 && p[m] > p[i2]) i2 = m;
      const float tv = p[i1] + p[i2];
#pragma unroll
      for (int m = 0; m < MMX; ++m)
        wmem[(bt + tid) * MMX + m] = (m == i1 || m == i2) ? p[m] / tv : 0.0f;
    }
    __syncthreads();

    {  // per-channel total decay accumulation (registers, channel = tid)
#pragma unroll
      for (int t = 0; t < TT; ++t) {
        const float g = gkls[t * DD + tid];
        const float* wp = wmem + (bt + t) * MMX;
        Gtot[0] += g;
#pragma unroll
        for (int e = 0; e < MMX; ++e) Gtot[e + 1] += (wp[e] > 0.0f) ? g : 0.0f;
      }
    }
    if (tile == NTILE - 1) {  // q_last = x[K-1,:] @ W_q (fp32, exact)
      const float* xr = xf + (TT - 1) * DD;
      float s = 0.0f;
#pragma unroll 8
      for (int d = 0; d < DD; ++d) s += xr[d] * W_q[d * DD + tid];
      qlast[tid] = s;
    }
    __syncthreads();
  }

  // =========================================================================
  // PASS 2: recompute k/v/gk tiles with WMMA; run the stable 5-memory scan.
  // =========================================================================
  float Grun[NMEM];
#pragma unroll
  for (int m = 0; m < NMEM; ++m) Grun[m] = 0.0f;
  float o_acc = 0.0f;
  const float ql = qlast[tid];
  const int h2 = (tid >> 6) * 2;  // waves covering this head

  for (int tile = 0; tile < NTILE; ++tile) {
    const int bt = tile * TT;
    {
      const float* kv = keyval + ((size_t)(b * KK + bt)) * DD;
      const float cv = cond[tid];
#pragma unroll
      for (int i = 0; i < TT; ++i)
        xbf[i * DD + tid] = f2bf(kv[i * DD + tid] + cv);
      if (tid < TT) mt[tid] = (float)amask[b * KK + bt + tid];
    }
    __syncthreads();

    if (wave == 0) {  // g1
      v8f acc = zero8();
#pragma unroll
      for (int kc = 0; kc < 8; ++kc) {
        v16bf a = lds_a_frag(xbf, lane, kc);
        v16bf bb = *(const v16bf*)(wsG1 + ((size_t)(kc * 32 + lane)) * 16);
        acc = wmma_bf16(a, bb, acc);
      }
      const int rh = ((lane >> 4) & 1) * 8;
#pragma unroll
      for (int r = 0; r < 8; ++r)
        g1bf[(r + rh) * 16 + (lane & 15)] = f2bf(acc[r]);
    }
    {  // k and v tiles: each wave owns 2 of 16 output column blocks
#pragma unroll
      for (int jj = 0; jj < 2; ++jj) {
        const int j = wave * 2 + jj;
        v8f ak = zero8(), av = zero8();
#pragma unroll
        for (int kc = 0; kc < 8; ++kc) {
          v16bf a = lds_a_frag(xbf, lane, kc);
          const __bf16* pk = wsWk + ((size_t)((kc * 16 + j) * 32 + lane)) * 16;
          const __bf16* pv = wsWv + ((size_t)((kc * 16 + j) * 32 + lane)) * 16;
          __builtin_prefetch(pk + 8192, 0, 0);  // next kc block
          __builtin_prefetch(pv + 8192, 0, 0);
          v16bf bk = *(const v16bf*)pk;
          v16bf bv = *(const v16bf*)pv;
          ak = wmma_bf16(a, bk, ak);
          av = wmma_bf16(a, bv, av);
        }
        const int nc = j * 16 + (lane & 15);
        const int rh = ((lane >> 4) & 1) * 8;
#pragma unroll
        for (int r = 0; r < 8; ++r) {
          const int tr = r + rh;
          const float m = mt[tr];
          kls[tr * DD + nc] = ak[r] * m;
          vls[tr * DD + nc] = av[r] * m;
        }
      }
    }
    __syncthreads();

    {  // gk tile
      v16bf ag = lds_a_frag_g1(g1bf, lane);
#pragma unroll
      for (int jj = 0; jj < 2; ++jj) {
        const int j = wave * 2 + jj;
        v16bf bb = *(const v16bf*)(wsG2 + ((size_t)(j * 32 + lane)) * 16);
        v8f acc = wmma_bf16(ag, bb, zero8());
        const int nc = j * 16 + (lane & 15);
        const int rh = ((lane >> 4) & 1) * 8;
#pragma unroll
        for (int r = 0; r < 8; ++r) {
          const int tr = r + rh;
          const float z = acc[r] + b_gk2[nc];
          gkls[tr * DD + nc] = log_sigmoid(z) * (1.0f / GN_F) * mt[tr];
        }
      }
    }
    __syncthreads();

    // ---- phase A: s_m[t,h] = sum_d q*k*sel*exp(Gtot - Grun) --------------
#pragma unroll 4
    for (int t = 0; t < TT; ++t) {
      const float g  = gkls[t * DD + tid];
      const float kq = ql * kls[t * DD + tid];
      const float* wp = wmem + (bt + t) * MMX;
      Grun[0] += g;
      float term = kq * __expf(Gtot[0] - Grun[0]);
      float rs = wave_reduce(term);
      if (lane == 0) spart[(t * NMEM + 0) * 8 + wave] = rs;
#pragma unroll
      for (int e = 0; e < MMX; ++e) {
        const bool sel = wp[e] > 0.0f;
        Grun[e + 1] += sel ? g : 0.0f;
        term = sel ? kq * __expf(Gtot[e + 1] - Grun[e + 1]) : 0.0f;
        rs = wave_reduce(term);
        if (lane == 0) spart[(t * NMEM + e + 1) * 8 + wave] = rs;
      }
    }
    __syncthreads();

    // ---- phase B: o[h,v] += (s0 + sum_e we*se) * v ------------------------
#pragma unroll 4
    for (int t = 0; t < TT; ++t) {
      const float sv = vls[t * DD + tid];
      const float* wp = wmem + (bt + t) * MMX;
      float acc = spart[(t * NMEM + 0) * 8 + h2] +
                  spart[(t * NMEM + 0) * 8 + h2 + 1];
#pragma unroll
      for (int e = 0; e < MMX; ++e) {
        const float se = spart[(t * NMEM + e + 1) * 8 + h2] +
                         spart[(t * NMEM + e + 1) * 8 + h2 + 1];
        acc += wp[e] * se;
      }
      o_acc += acc * sv;
    }
    __syncthreads();
  }

  // ---- RMS norm over DV per head, then o_flat @ W_o -----------------------
  {
    float rs = wave_reduce(o_acc * o_acc);
    if (lane == 0) red[wave] = rs;
  }
  __syncthreads();
  {
    const float ms = (red[h2] + red[h2 + 1]) * (1.0f / (float)DVV);
    oflat[tid] = o_acc * rsqrtf(ms + EPS_F) * norm_w[tid & (DVV - 1)];
  }
  __syncthreads();
  {
    float s = 0.0f;
#pragma unroll 8
    for (int c = 0; c < HH * DVV; ++c) s += oflat[c] * W_o[c * DD + tid];
    out[(size_t)n * DD + tid] = s;
  }
}

// ---------------------------------------------------------------------------
extern "C" void kernel_launch(void* const* d_in, const int* in_sizes, int n_in,
                              void* d_out, int out_size, void* d_ws,
                              size_t ws_size, hipStream_t stream) {
  const float* query    = (const float*)d_in[0];
  const float* keyval   = (const float*)d_in[1];
  const int*   amask    = (const int*)d_in[2];
  const float* W_cond   = (const float*)d_in[3];
  const float* W_q      = (const float*)d_in[4];
  const float* W_k      = (const float*)d_in[5];
  const float* W_v      = (const float*)d_in[6];
  const float* W_gk1    = (const float*)d_in[7];
  const float* W_gk2    = (const float*)d_in[8];
  const float* b_gk2    = (const float*)d_in[9];
  const float* W_router = (const float*)d_in[10];
  const float* norm_w   = (const float*)d_in[11];
  const float* W_o      = (const float*)d_in[12];
  (void)in_sizes; (void)n_in; (void)out_size; (void)ws_size;

  float* out = (float*)d_out;                  // (B,Q,D) fp32
  float* rlogits = out + (size_t)NN * DD;      // (N*K, M) fp32
  __bf16* ws = (__bf16*)d_ws;

  swizzle_weights<<<(WS_TOT + 255) / 256, 256, 0, stream>>>(
      W_k, W_v, W_gk1, W_gk2, ws);
  mom_attn_kernel<<<NN, 256, MAIN_SMEM, stream>>>(
      query, keyval, amask, W_cond, W_q, b_gk2, W_router, norm_w, W_o, ws,
      out, rlogits);
}
